// NoisyTopKRouter_19688130084977
// MI455X (gfx1250) — matrix operand order repaired and checked by reading, not compiled, not run on hardware
//
#include <hip/hip_runtime.h>
#include <hip/hip_bf16.h>

typedef __bf16 bf16_t;
typedef __attribute__((ext_vector_type(16))) __bf16 v16bf;
typedef __attribute__((ext_vector_type(8)))  float  v8f;
typedef int v4i __attribute__((vector_size(16)));   // <4 x i32>, matches async builtin pointee

constexpr int KT = 32;          // WMMA bf16 K per step
constexpr int LP = KT + 8;      // padded LDS row pitch (40 bf16 = 80B: 16B-aligned, conflict-free)
constexpr int NTOK = 32768;
constexpr int DIN  = 1024;
constexpr int H2   = 2048;
constexpr int HID  = 1024;
constexpr int NE   = 64;
constexpr int TOPK = 8;

// ---------------- CDNA5 async global->LDS copy (ASYNCcnt path), with fallback
__device__ __forceinline__ void async_copy_b128(const bf16_t* g, bf16_t* l) {
#if defined(__gfx1250__) && __has_builtin(__builtin_amdgcn_global_load_async_to_lds_b128)
  typedef __attribute__((address_space(1))) v4i* gp_t;   // global <4 x i32>*
  typedef __attribute__((address_space(3))) v4i* lp_t;   // LDS    <4 x i32>*
  __builtin_amdgcn_global_load_async_to_lds_b128((gp_t)g, (lp_t)l, 0, 0);
#else
  *reinterpret_cast<uint4*>(l) = *reinterpret_cast<const uint4*>(g);
#endif
}
__device__ __forceinline__ void wait_async_lds() {
#if defined(__gfx1250__) && __has_builtin(__builtin_amdgcn_s_wait_asynccnt)
  __builtin_amdgcn_s_wait_asynccnt(0);
#elif defined(__gfx1250__)
  asm volatile("s_wait_asynccnt 0" ::: "memory");
#endif
}

// ---------------------------------------------------------------- f32 -> bf16
__global__ __launch_bounds__(256)
void f32_to_bf16_kernel(const float* __restrict__ in, bf16_t* __restrict__ out, size_t n) {
  size_t i = (size_t)blockIdx.x * 256 + threadIdx.x;
  if (i < n) out[i] = (bf16_t)in[i];
}

// ---------------------------------------------------------------- WMMA GEMM
// C[M, Ncols] = act(A[M,K] @ W[K,Ncols] + bias), A/W bf16 row-major, f32 accum.
// 256 threads = 8 wave32. Block tile BM x BN; each wave computes 32 x (BN/2).
// A staged row-major (async global->LDS); B staged transposed so both A and B
// fragments are pure 128-bit LDS loads in the inner loop.
template<int BM, int BN, bool RELU, bool OUT_BF16>
__global__ __launch_bounds__(256)
void gemm_bias_act(const bf16_t* __restrict__ A, const bf16_t* __restrict__ W,
                   const float* __restrict__ bias,
                   bf16_t* __restrict__ outB, float* __restrict__ outF,
                   int M, int Ncols, int K)
{
  constexpr int WROWS = 4, WCOLS = 2;
  constexpr int WM = BM / WROWS;   // 32
  constexpr int WN = BN / WCOLS;   // 64 (BN=128) or 32 (BN=64)
  constexpr int MT = WM / 16;      // 2
  constexpr int NT = WN / 16;      // 4 or 2

  __shared__ bf16_t As [BM][LP];   // A tile, row-major along K
  __shared__ bf16_t BsT[BN][LP];   // B tile, transposed: BsT[n][k]

  const int tid  = threadIdx.x;
  const int wave = tid >> 5;
  const int lane = tid & 31;
  const int wrow = wave / WCOLS;
  const int wcol = wave % WCOLS;
  const int lhalf = lane >> 4;     // 0: lanes 0-15, 1: lanes 16-31
  const int lmod  = lane & 15;

  const int block_m = blockIdx.y * BM;
  const int block_n = blockIdx.x * BN;

  v8f acc[MT][NT];
#pragma unroll
  for (int i = 0; i < MT; ++i)
#pragma unroll
    for (int j = 0; j < NT; ++j)
      acc[i][j] = v8f{0.f,0.f,0.f,0.f,0.f,0.f,0.f,0.f};

  for (int kk = 0; kk < K; kk += KT) {
    __syncthreads();   // previous-iteration LDS reads complete before overwrite

    // ---- stage A tile (BM x KT): 128-bit async copies global->LDS
    constexpr int A_VECS = BM * KT / 8;
#pragma unroll
    for (int idx = tid; idx < A_VECS; idx += 256) {
      const int off = idx * 8;
      const int r = off / KT, c = off % KT;
      async_copy_b128(&A[(size_t)(block_m + r) * K + kk + c], &As[r][c]);
    }
    // ---- stage B tile (KT x BN) transposed into BsT[n][k]
    constexpr int B_VECS = KT * BN / 8;
#pragma unroll
    for (int idx = tid; idx < B_VECS; idx += 256) {
      const int off = idx * 8;
      const int r = off / BN, c = off % BN;     // r = k row, c = n col
      const size_t g = (size_t)(kk + r) * Ncols + block_n + c;
      uint4 wv = *reinterpret_cast<const uint4*>(&W[g]);
      if (kk + KT < K) __builtin_prefetch(&W[g + (size_t)KT * Ncols], 0, 0);
      const bf16_t* w8 = reinterpret_cast<const bf16_t*>(&wv);
#pragma unroll
      for (int e = 0; e < 8; ++e) BsT[c + e][r] = w8[e];
    }
    wait_async_lds();
    __syncthreads();

    // ---- WMMA inner loop: all fragment gathers are ds_load_b128
#pragma unroll
    for (int i = 0; i < MT; ++i) {
      union { v16bf v; uint4 u[2]; } af;
      const int m = wrow * WM + i * 16 + lmod;
      // A frag (ISA 16-bit A 16x32 layout): lanes 0-15 K=0-7|16-23, lanes 16-31 K=8-15|24-31
      af.u[0] = *reinterpret_cast<const uint4*>(&As[m][lhalf * 8]);
      af.u[1] = *reinterpret_cast<const uint4*>(&As[m][16 + lhalf * 8]);
#pragma unroll
      for (int j = 0; j < NT; ++j) {
        union { v16bf v; uint4 u[2]; } bf;
        const int n = wcol * WN + j * 16 + lmod;
        // B frag: lane's column n, K = lhalf*8 .. +7 | 16+lhalf*8 .. +7 (matches A)
        bf.u[0] = *reinterpret_cast<const uint4*>(&BsT[n][lhalf * 8]);
        bf.u[1] = *reinterpret_cast<const uint4*>(&BsT[n][16 + lhalf * 8]);
        acc[i][j] = __builtin_amdgcn_wmma_f32_16x16x32_bf16(
            /*neg_a=*/false, af.v, /*neg_b=*/false, bf.v,
            /*c_mod=*/(short)0, acc[i][j], /*reuse_a=*/false, /*reuse_b=*/false);
      }
    }
  }

  // ---- epilogue: D layout (vgpr r, lanes>=16 -> M = 8+r), bias + act + store
#pragma unroll
  for (int i = 0; i < MT; ++i) {
#pragma unroll
    for (int j = 0; j < NT; ++j) {
      const int n = block_n + wcol * WN + j * 16 + lmod;
      const float bv = bias[n];
#pragma unroll
      for (int r = 0; r < 8; ++r) {
        const int m = block_m + wrow * WM + i * 16 + lhalf * 8 + r;
        float v = acc[i][j][r] + bv;
        if (RELU) v = v > 0.f ? v : 0.f;
        if (OUT_BF16) outB[(size_t)m * Ncols + n] = (bf16_t)v;
        else          outF[(size_t)m * Ncols + n] = v;
      }
    }
  }
}

// ------------------------------------------------- noisy top-k + sparse softmax
__global__ __launch_bounds__(256)
void router_kernel(const float* __restrict__ logits, const float* __restrict__ nlogits,
                   const float* __restrict__ noise,
                   float* __restrict__ out, int* __restrict__ idx_out, int ntok)
{
  const int t = blockIdx.x * 256 + threadIdx.x;
  if (t >= ntok) return;

  float v[NE];
#pragma unroll 4
  for (int e = 0; e < NE; ++e) {
    const float nl = nlogits[(size_t)t * NE + e];
    // stable softplus: max(x,0) + log1p(exp(-|x|))
    const float sp = fmaxf(nl, 0.f) + log1pf(__expf(-fabsf(nl)));
    v[e] = logits[(size_t)t * NE + e] + noise[(size_t)t * NE + e] * sp;
  }

  int   sel[TOPK];
  float selv[TOPK];
  unsigned long long taken = 0ull;
#pragma unroll
  for (int k = 0; k < TOPK; ++k) {
    float best = -INFINITY; int bi = 0;
    for (int e = 0; e < NE; ++e) {
      const bool free_slot = ((taken >> e) & 1ull) == 0ull;
      if (free_slot && v[e] > best) { best = v[e]; bi = e; }
    }
    taken |= (1ull << bi);
    sel[k] = bi; selv[k] = best;
  }

  const float mx = selv[0];
  float ex[TOPK], sum = 0.f;
#pragma unroll
  for (int k = 0; k < TOPK; ++k) { ex[k] = __expf(selv[k] - mx); sum += ex[k]; }
  const float inv = 1.f / sum;

  float o[NE];
#pragma unroll 4
  for (int e = 0; e < NE; ++e) o[e] = 0.f;
#pragma unroll
  for (int k = 0; k < TOPK; ++k) o[sel[k]] = ex[k] * inv;
#pragma unroll 4
  for (int e = 0; e < NE; ++e) out[(size_t)t * NE + e] = o[e];
#pragma unroll
  for (int k = 0; k < TOPK; ++k) idx_out[(size_t)t * TOPK + k] = sel[k];
}

// ---------------------------------------------------------------- launcher
static inline size_t align256(size_t x) { return (x + 255) & ~(size_t)255; }

extern "C" void kernel_launch(void* const* d_in, const int* in_sizes, int n_in,
                              void* d_out, int out_size, void* d_ws, size_t ws_size,
                              hipStream_t stream) {
  (void)in_sizes; (void)n_in; (void)out_size; (void)ws_size;
  const float* x     = (const float*)d_in[0];
  const float* noise = (const float*)d_in[1];
  const float* w1 = (const float*)d_in[2];  const float* b1 = (const float*)d_in[3];
  const float* w2 = (const float*)d_in[4];  const float* b2 = (const float*)d_in[5];
  const float* wn = (const float*)d_in[6];  const float* bn = (const float*)d_in[7];
  const float* w3 = (const float*)d_in[8];  const float* b3 = (const float*)d_in[9];
  const float* w4 = (const float*)d_in[10]; const float* b4 = (const float*)d_in[11];
  const float* wz = (const float*)d_in[12]; const float* bz = (const float*)d_in[13];

  // workspace carve-up
  char* p = (char*)d_ws;
  auto carve = [&](size_t bytes) { char* r = p; p += align256(bytes); return r; };
  bf16_t* xb  = (bf16_t*)carve((size_t)NTOK * DIN * 2);
  bf16_t* hA  = (bf16_t*)carve((size_t)NTOK * H2  * 2);
  bf16_t* hB  = (bf16_t*)carve((size_t)NTOK * H2  * 2);
  bf16_t* w1b = (bf16_t*)carve((size_t)DIN * H2  * 2);
  bf16_t* w2b = (bf16_t*)carve((size_t)H2  * H2  * 2);
  bf16_t* wnb = (bf16_t*)carve((size_t)H2  * H2  * 2);
  bf16_t* w3b = (bf16_t*)carve((size_t)H2  * HID * 2);
  bf16_t* w4b = (bf16_t*)carve((size_t)HID * NE  * 2);
  bf16_t* wzb = (bf16_t*)carve((size_t)DIN * NE  * 2);
  float*  lg  = (float*) carve((size_t)NTOK * NE * 4);
  float*  nlg = (float*) carve((size_t)NTOK * NE * 4);

  auto cvt = [&](const float* src, bf16_t* dst, size_t n) {
    f32_to_bf16_kernel<<<dim3((unsigned)((n + 255) / 256)), 256, 0, stream>>>(src, dst, n);
  };
  cvt(x,  xb,  (size_t)NTOK * DIN);
  cvt(w1, w1b, (size_t)DIN * H2);
  cvt(w2, w2b, (size_t)H2 * H2);
  cvt(wn, wnb, (size_t)H2 * H2);
  cvt(w3, w3b, (size_t)H2 * HID);
  cvt(w4, w4b, (size_t)HID * NE);
  cvt(wz, wzb, (size_t)DIN * NE);

  // layer chain (ping-pong hA/hB; xb preserved for noise projection)
  gemm_bias_act<128,128,true,true ><<<dim3(H2/128,  NTOK/128), 256, 0, stream>>>(xb, w1b, b1, hA, nullptr, NTOK, H2,  DIN);
  gemm_bias_act<128,128,true,true ><<<dim3(H2/128,  NTOK/128), 256, 0, stream>>>(hA, w2b, b2, hB, nullptr, NTOK, H2,  H2);
  gemm_bias_act<128,128,true,true ><<<dim3(H2/128,  NTOK/128), 256, 0, stream>>>(hB, wnb, bn, hA, nullptr, NTOK, H2,  H2);
  gemm_bias_act<128,128,true,true ><<<dim3(HID/128, NTOK/128), 256, 0, stream>>>(hA, w3b, b3, hB, nullptr, NTOK, HID, H2);
  gemm_bias_act<128,64,false,false><<<dim3(NE/64,   NTOK/128), 256, 0, stream>>>(hB, w4b, b4, nullptr, lg,  NTOK, NE,  HID);
  gemm_bias_act<128,64,false,false><<<dim3(NE/64,   NTOK/128), 256, 0, stream>>>(xb, wzb, bz, nullptr, nlg, NTOK, NE,  DIN);

  float* out_router = (float*)d_out;
  int*   out_idx    = (int*)(out_router + (size_t)NTOK * NE);
  router_kernel<<<dim3(NTOK / 256), 256, 0, stream>>>(lg, nlg, noise, out_router, out_idx, NTOK);
}
